// SelfAttention_17377437680246
// MI455X (gfx1250) — matrix operand order repaired and checked by reading, not compiled, not run on hardware
//
#include <hip/hip_runtime.h>
#include <hip/hip_bf16.h>

// ---------------------------------------------------------------------------
// SelfAttention (diagonal-softmax variant) for MI455X / gfx1250, wave32 WMMA.
// N=4, L=4096, E=1024, H=1024.
//   Q = X Wq^T + bq ; K = X Wk^T + bk ; V = X Wv^T + bv
//   S = Q K^T / sqrt(L); A = softmax over query axis; out = diag(A) * V
// Only per-column (max, sumexp) of S and S's diagonal are needed -> S is
// never materialized (online reduction, flash-attention style).
// ---------------------------------------------------------------------------

typedef __attribute__((ext_vector_type(16))) __bf16 v16bf;
typedef __attribute__((ext_vector_type(8)))  float  v8f;

#define NB 4
#define LL 4096
#define EE 1024
#define HH 1024
#define LTILES (LL / 16)          // 256 column tiles per batch
#define SCALE 0.015625f           // 1/sqrt(4096)

static __device__ __forceinline__ v8f wmma_bf16(v16bf a, v16bf b, v8f c) {
  // D = A(16x32 bf16) * B(32x16 bf16) + C(16x16 f32)
  return __builtin_amdgcn_wmma_f32_16x16x32_bf16(
      /*neg_a=*/false, a, /*neg_b=*/false, b,
      /*c_mod=*/(short)0, c, /*reuse_a=*/false, /*reuse_b=*/false);
}

// ---------------------------------------------------------------------------
// Kernel 1: fused QKV projection. One A (X) fragment feeds 3 WMMAs per
// 32-wide K step. Q,K written bf16 (feed the score GEMM), V written f32.
// grid = (H/16, (N*L)/(16*8)), block = 256 (8 waves), wave tile = 16x16.
// ---------------------------------------------------------------------------
__global__ __launch_bounds__(256) void qkv_gemm_kernel(
    const float* __restrict__ X,
    const float* __restrict__ Wq, const float* __restrict__ bq,
    const float* __restrict__ Wk, const float* __restrict__ bk,
    const float* __restrict__ Wv, const float* __restrict__ bv,
    __bf16* __restrict__ Qb, __bf16* __restrict__ Kb, float* __restrict__ Vf)
{
  const int lane  = threadIdx.x & 31;
  const int wave  = threadIdx.x >> 5;
  const int h0    = blockIdx.x * 16;
  const int row0  = (blockIdx.y * 8 + wave) * 16;

  const int nlo    = lane & 15;   // A: row within tile; B/D: column within tile
  const int khi    = lane >> 4;   // lane half selector
  const int kbaseA = khi * 8;     // unused; A loads also use kbaseB
  const int kbaseB = khi * 16;    // B fragment K sub-base (contiguous k = 16*half + s, shared with A)

  const float* Xr  = X  + (size_t)(row0 + nlo) * EE;
  const float* Wqr = Wq + (size_t)(h0   + nlo) * EE;   // B[k,n] = W[h0+n, k]
  const float* Wkr = Wk + (size_t)(h0   + nlo) * EE;
  const float* Wvr = Wv + (size_t)(h0   + nlo) * EE;

  v8f aq = {}, ak = {}, av = {};

  for (int kk = 0; kk < EE; kk += 32) {
    v16bf fa, fq, fk, fv;
#pragma unroll
    for (int i = 0; i < 8; ++i) {
      // A layout: lanes 0-15 hold K=0..15, lanes 16-31 hold K=16..31 (contiguous, matches B)
      const int ka = kk + kbaseB + 2 * i;
      fa[2 * i]     = (__bf16)Xr[ka];
      fa[2 * i + 1] = (__bf16)Xr[ka + 1];
      // B layout: lanes 0-15 hold K=0..15, lanes 16-31 hold K=16..31 (contiguous)
      const int kb = kk + kbaseB + 2 * i;
      fq[2 * i] = (__bf16)Wqr[kb]; fq[2 * i + 1] = (__bf16)Wqr[kb + 1];
      fk[2 * i] = (__bf16)Wkr[kb]; fk[2 * i + 1] = (__bf16)Wkr[kb + 1];
      fv[2 * i] = (__bf16)Wvr[kb]; fv[2 * i + 1] = (__bf16)Wvr[kb + 1];
    }
    aq = wmma_bf16(fa, fq, aq);
    ak = wmma_bf16(fa, fk, ak);
    av = wmma_bf16(fa, fv, av);
  }

  const float biasq = bq[h0 + nlo];
  const float biask = bk[h0 + nlo];
  const float biasv = bv[h0 + nlo];

#pragma unroll
  for (int v = 0; v < 8; ++v) {
    const int m = khi * 8 + v;                     // C/D layout row
    const size_t idx = (size_t)(row0 + m) * HH + (h0 + nlo);
    Qb[idx] = (__bf16)(aq[v] + biasq);
    Kb[idx] = (__bf16)(ak[v] + biask);
    Vf[idx] = av[v] + biasv;
  }
}

// ---------------------------------------------------------------------------
// Kernel 2: S = Q K^T * SCALE, online per-column (max, sumexp) + diagonal.
// grid = (N, LTILES): one block per (batch, 16-column strip).
// 8 waves split the 4096 query rows (512 each, 32 tiles in groups of 4 so the
// K fragment is reused across 4 WMMAs). S is never written to memory.
// ---------------------------------------------------------------------------
__global__ __launch_bounds__(256) void scores_kernel(
    const __bf16* __restrict__ Qb, const __bf16* __restrict__ Kb,
    float2* __restrict__ partial, float* __restrict__ diagraw)
{
  const int lane  = threadIdx.x & 31;
  const int wave  = threadIdx.x >> 5;
  const int n     = blockIdx.x;
  const int btile = blockIdx.y;
  const int b0    = btile * 16;

  const int nlo    = lane & 15;
  const int khi    = lane >> 4;
  const int kbaseA = khi * 8;
  const int kbaseB = khi * 16;

  const __bf16* Qn = Qb + (size_t)n * LL * HH;
  const __bf16* Kr = Kb + (size_t)n * LL * HH + (size_t)(b0 + nlo) * HH;

  float mx = -3.402823466e38f;   // running max (this lane's rows of its column)
  float sm = 0.f;                // running sum of exp

  const int vdiag = nlo - khi * 8;   // which D element is on the diagonal (if 0..7)

  for (int g = 0; g < 8; ++g) {
    const int t0 = wave * 32 + g * 4;     // first query tile of this group
    v8f acc[4] = {{}, {}, {}, {}};

    for (int kk = 0; kk < HH; kk += 32) {
      v16bf fb;                            // B = K rows (shared by 4 tiles)
#pragma unroll
      for (int i = 0; i < 16; ++i) fb[i] = Kr[kk + kbaseB + i];
#pragma unroll
      for (int j = 0; j < 4; ++j) {
        const __bf16* Qr = Qn + (size_t)((t0 + j) * 16 + nlo) * HH;
        v16bf fa;
#pragma unroll
        for (int i = 0; i < 8; ++i) {
          const int ka = kk + kbaseB + 2 * i;
          fa[2 * i]     = Qr[ka];
          fa[2 * i + 1] = Qr[ka + 1];
        }
        acc[j] = wmma_bf16(fa, fb, acc[j]);
      }
    }

#pragma unroll
    for (int j = 0; j < 4; ++j) {
      const bool isDiag = ((t0 + j) == btile);
#pragma unroll
      for (int v = 0; v < 8; ++v) {
        const float val = acc[j][v] * SCALE;
        const float nm  = fmaxf(mx, val);
        sm = sm * __expf(mx - nm) + __expf(val - nm);
        mx = nm;
        if (isDiag && v == vdiag)
          diagraw[(size_t)n * LL + b0 + nlo] = val;   // raw scaled S[b,b]
      }
    }
  }

  // Merge the two lane-halves of each column (rows m and m+8).
  const float om = __shfl_xor(mx, 16, 32);
  const float os = __shfl_xor(sm, 16, 32);
  const float M  = fmaxf(mx, om);
  const float S  = sm * __expf(mx - M) + os * __expf(om - M);

  if (lane < 16)
    partial[(((size_t)n * LTILES + btile) * 16 + lane) * 8 + wave] =
        make_float2(M, S);
}

// ---------------------------------------------------------------------------
// Kernel 3: combine 8 wave-partials per column, diagonal softmax weight,
// out[n,l,:] = w * V[n,l,:].  grid = N*L blocks, 256 threads (float4 each).
// ---------------------------------------------------------------------------
__global__ __launch_bounds__(256) void finalize_kernel(
    const float* __restrict__ Vf, const float* __restrict__ diagraw,
    const float2* __restrict__ partial, float* __restrict__ out)
{
  const int r = blockIdx.x;            // flattened (n,l)
  const int n = r >> 12;               // / 4096
  const int l = r & (LL - 1);

  __shared__ float sdval;
  if (threadIdx.x == 0) {
    const int btile = l >> 4, col = l & 15;
    const float2* p = partial + (((size_t)n * LTILES + btile) * 16 + col) * 8;
    float M = -3.402823466e38f;
#pragma unroll
    for (int i = 0; i < 8; ++i) M = fmaxf(M, p[i].x);
    float S = 0.f;
#pragma unroll
    for (int i = 0; i < 8; ++i) S += p[i].y * __expf(p[i].x - M);
    sdval = __expf(diagraw[(size_t)n * LL + l] - M) / S;
  }
  __syncthreads();

  const float d = sdval;
  const float4 vv = ((const float4*)(Vf + (size_t)r * HH))[threadIdx.x];
  float4 o;
  o.x = vv.x * d; o.y = vv.y * d; o.z = vv.z * d; o.w = vv.w * d;
  ((float4*)(out + (size_t)r * HH))[threadIdx.x] = o;
}

// ---------------------------------------------------------------------------
extern "C" void kernel_launch(void* const* d_in, const int* in_sizes, int n_in,
                              void* d_out, int out_size, void* d_ws, size_t ws_size,
                              hipStream_t stream) {
  const float* X  = (const float*)d_in[0];
  const float* Wq = (const float*)d_in[1];
  const float* bq = (const float*)d_in[2];
  const float* Wk = (const float*)d_in[3];
  const float* bk = (const float*)d_in[4];
  const float* Wv = (const float*)d_in[5];
  const float* bv = (const float*)d_in[6];

  char* ws = (char*)d_ws;
  const size_t qBytes = (size_t)NB * LL * HH * sizeof(__bf16);           // 32 MiB
  const size_t vBytes = (size_t)NB * LL * HH * sizeof(float);            // 64 MiB
  const size_t pBytes = (size_t)NB * LTILES * 16 * 8 * sizeof(float2);   // 1 MiB

  __bf16* Qb      = (__bf16*)(ws);
  __bf16* Kb      = (__bf16*)(ws + qBytes);
  float*  Vf      = (float*) (ws + 2 * qBytes);
  float2* partial = (float2*)(ws + 2 * qBytes + vBytes);
  float*  diagraw = (float*) (ws + 2 * qBytes + vBytes + pBytes);

  const dim3 blk(256);
  qkv_gemm_kernel<<<dim3(HH / 16, (NB * LL) / (16 * 8)), blk, 0, stream>>>(
      X, Wq, bq, Wk, bk, Wv, bv, Qb, Kb, Vf);
  scores_kernel<<<dim3(NB, LTILES), blk, 0, stream>>>(Qb, Kb, partial, diagraw);
  finalize_kernel<<<dim3(NB * LL), blk, 0, stream>>>(Vf, diagraw, partial,
                                                     (float*)d_out);
}
